// NodeUpdateBlock_9972914061334
// MI455X (gfx1250) — matrix operand order repaired and checked by reading, not compiled, not run on hardware
//
#include <hip/hip_runtime.h>
#include <math.h>

// ---------------------------------------------------------------------------
// Types for CDNA5 WMMA (wave32, 16x16x32 f16 -> f32)
// ---------------------------------------------------------------------------
typedef __attribute__((ext_vector_type(16))) _Float16 v16h;
typedef __attribute__((ext_vector_type(8)))  _Float16 v8h;
typedef __attribute__((ext_vector_type(8)))  float    v8f;
typedef __attribute__((ext_vector_type(8)))  unsigned v8u;

constexpr int kN    = 16000;
constexpr int kE    = 256000;
constexpr int kDIM  = 160;   // 64 scalar + 32*3 vector

// ---------------------------------------------------------------------------
// Weight fragment area: each 32(K)x16(N) f16 tile = 256 u32, lane-major:
//   word[lane*8 + j] packs W[k][n], W[k+1][n] with k = kt*32 + 2j + 16*(lane>=16),
//   n = nt*16 + (lane&15).  Tile index = nt*KT + kt.
constexpr int OFF_WSS   = 0;      // K192 N96 -> 36 tiles
constexpr int OFF_WVV   = 9216;   // K96  N96 -> 18
constexpr int OFF_WSV   = 13824;  // K192 N32 -> 12
constexpr int OFF_WVS   = 16896;  // K96  N32 -> 6
constexpr int OFF_FC0   = 18432;  // K64  N64 -> 8
constexpr int OFF_FC1   = 20480;  // 8
constexpr int OFF_FC2   = 22528;  // K64  N96 -> 12
constexpr int OFF_PRES  = 25600;  // K64  N64 -> 8
constexpr int OFF_PREV  = 27648;  // K32  N32 -> 2
constexpr int OFF_POSTS = 28160;  // 8
constexpr int OFF_POSTV = 30208;  // 2
constexpr int OFF_SCS   = 30720;  // K256 N64 -> 32
constexpr int OFF_SCV   = 38912;  // K128 N32 -> 8
constexpr int WTS_WORDS = 40960;  // 160 KB

// ---------------------------------------------------------------------------
// Workspace layout (bytes)
// ---------------------------------------------------------------------------
constexpr size_t WOFF_X16   = (size_t)WTS_WORDS * 4;              // f16 node feats
constexpr size_t WOFF_SC    = WOFF_X16 + (size_t)kN * kDIM * 2;   // f32 self-connection
constexpr size_t WOFF_ACC   = WOFF_SC  + (size_t)kN * kDIM * 4;   // f32 scatter accumulator
constexpr size_t WOFF_CNT   = WOFF_ACC + (size_t)kN * kDIM * 4;   // f32 edge counts
constexpr size_t WOFF_PRE   = WOFF_CNT + (size_t)kN * 4;          // f32 pre-layernorm
constexpr size_t WOFF_STATS = WOFF_PRE + (size_t)kN * kDIM * 4;   // 64 floats batch stats
// stats: [0..3] cntb, [4..7] ssum, [8..19] vsum(b*3+d), [20..23] vars, [24..27] varv

// ---------------------------------------------------------------------------
// Helpers
// ---------------------------------------------------------------------------
__device__ __forceinline__ v8f wmma16(v16h a, v16h b, v8f c) {
  return __builtin_amdgcn_wmma_f32_16x16x32_f16(false, a, false, b, (short)0, c,
                                                false, false);
}

// relaxed agent-scope f32 atomic add -> native global_atomic_add_f32
__device__ __forceinline__ void atomicAddF(float* p, float v) {
  __hip_atomic_fetch_add(p, v, __ATOMIC_RELAXED, __HIP_MEMORY_SCOPE_AGENT);
}

// B fragment: pre-swizzled weights, two coalesced 16B loads per lane.
__device__ __forceinline__ v16h load_b(const unsigned* __restrict__ wbase, int tile) {
  const unsigned* p = wbase + (tile << 8) + ((threadIdx.x & 31) << 3);
  v8u u;
#pragma unroll
  for (int j = 0; j < 8; ++j) u[j] = p[j];
  return __builtin_bit_cast(v16h, u);
}

// A fragment from LDS row-major [16][ldk] f16 tile, K window at k0 (even).
__device__ __forceinline__ v16h load_a(const _Float16* __restrict__ lds, int ldk, int k0) {
  int lane = threadIdx.x & 31;
  int M = lane & 15, hi = lane >> 4;
  const unsigned* row = (const unsigned*)(lds + M * ldk + k0);
  v8u u;
#pragma unroll
  for (int j = 0; j < 8; ++j) {
    int k = 2 * j + 16 * hi;
    u[j] = row[k >> 1];
  }
  return __builtin_bit_cast(v16h, u);
}

__device__ __forceinline__ float siluf(float x) { return x / (1.0f + __expf(-x)); }

__device__ __forceinline__ v8h cvt8(float4 a, float4 b) {
  v8h h;
  h[0] = (_Float16)a.x; h[1] = (_Float16)a.y; h[2] = (_Float16)a.z; h[3] = (_Float16)a.w;
  h[4] = (_Float16)b.x; h[5] = (_Float16)b.y; h[6] = (_Float16)b.z; h[7] = (_Float16)b.w;
  return h;
}

// ---------------------------------------------------------------------------
// Kernel 0: f32 weight matrix -> f16 B-fragment tiles (scale folded)
// element(k,n) = transposed ? W[n*ld + k] : W[k*ld + n]
// ---------------------------------------------------------------------------
__global__ void prep_weights(const float* __restrict__ W, unsigned* __restrict__ out,
                             int K, int Nn, int ld, int transposed, float scale) {
  int tiles_k = K >> 5, tiles_n = Nn >> 4;
  int total = tiles_k * tiles_n * 256;
  for (int w = blockIdx.x * blockDim.x + threadIdx.x; w < total;
       w += gridDim.x * blockDim.x) {
    int tile = w >> 8, lw = w & 255;
    int lane = lw >> 3, j = lw & 7;
    int nt = tile / tiles_k, kt = tile - nt * tiles_k;
    int nl = lane & 15, hi = lane >> 4;
    int k = kt * 32 + 2 * j + 16 * hi;
    int n = nt * 16 + nl;
    float w0, w1;
    if (transposed) { w0 = W[n * ld + k];  w1 = W[n * ld + k + 1]; }
    else            { w0 = W[k * ld + n];  w1 = W[(k + 1) * ld + n]; }
    _Float16 h0 = (_Float16)(w0 * scale), h1 = (_Float16)(w1 * scale);
    unsigned u0 = __builtin_bit_cast(unsigned short, h0);
    unsigned u1 = __builtin_bit_cast(unsigned short, h1);
    out[(tile << 8) + lane * 8 + j] = u0 | (u1 << 16);
  }
}

// ---------------------------------------------------------------------------
// Kernel 1: node preprocessing (x = e3linear(node_fea,Wpre) -> f16; sc -> f32)
// ---------------------------------------------------------------------------
__global__ void __launch_bounds__(32)
node_pre_kernel(const float* __restrict__ node_fea, const float* __restrict__ one_hot,
                const unsigned* __restrict__ wts, _Float16* __restrict__ x16,
                float* __restrict__ sc) {
  __shared__ _Float16 sfeat[16 * 64];
  __shared__ _Float16 vfeat[3][16 * 32];
  __shared__ _Float16 sexp[16 * 256];
  __shared__ _Float16 vexp[3][16 * 128];
  __shared__ int kspec[16];

  int lane = threadIdx.x;
  int n0 = blockIdx.x * 16;

  if (lane < 16) {
    const float* oh = one_hot + (size_t)(n0 + lane) * 4;
    int k = 0; float best = oh[0];
#pragma unroll
    for (int c = 1; c < 4; ++c) { if (oh[c] > best) { best = oh[c]; k = c; } }
    kspec[lane] = k;
  }
  for (int t = lane; t < 2048; t += 32) ((unsigned*)sexp)[t] = 0u;
  for (int t = lane; t < 3072; t += 32) ((unsigned*)vexp)[t] = 0u;
  __syncthreads();

  for (int t = lane; t < 16 * 160; t += 32) {
    int m = t / 160, c = t - m * 160;
    float val = node_fea[(size_t)(n0 + m) * kDIM + c];
    int k = kspec[m];
    _Float16 h = (_Float16)val;
    if (c < 64) {
      sfeat[m * 64 + c] = h;
      sexp[m * 256 + c * 4 + k] = h;
    } else {
      int i = (c - 64) / 3, d = (c - 64) - 3 * i;
      vfeat[d][m * 32 + i] = h;
      vexp[d][m * 128 + i * 4 + k] = h;
    }
  }
  __syncthreads();

  // K-outer / N-inner: each A fragment loaded once.
  v8f xs[4] = {}, xv[6] = {}, scs[4] = {}, scv[6] = {};
#pragma unroll
  for (int kt = 0; kt < 2; ++kt) {
    v16h a = load_a(sfeat, 64, kt * 32);
#pragma unroll
    for (int nt = 0; nt < 4; ++nt)
      xs[nt] = wmma16(a, load_b(wts + OFF_PRES, nt * 2 + kt), xs[nt]);
  }
#pragma unroll
  for (int d = 0; d < 3; ++d) {
    v16h a = load_a(vfeat[d], 32, 0);
#pragma unroll
    for (int nt = 0; nt < 2; ++nt)
      xv[d * 2 + nt] = wmma16(a, load_b(wts + OFF_PREV, nt), xv[d * 2 + nt]);
  }
#pragma unroll
  for (int kt = 0; kt < 8; ++kt) {
    v16h a = load_a(sexp, 256, kt * 32);
#pragma unroll
    for (int nt = 0; nt < 4; ++nt)
      scs[nt] = wmma16(a, load_b(wts + OFF_SCS, nt * 8 + kt), scs[nt]);
  }
#pragma unroll
  for (int d = 0; d < 3; ++d)
#pragma unroll
    for (int kt = 0; kt < 4; ++kt) {
      v16h a = load_a(vexp[d], 128, kt * 32);
#pragma unroll
      for (int nt = 0; nt < 2; ++nt)
        scv[d * 2 + nt] = wmma16(a, load_b(wts + OFF_SCV, nt * 4 + kt), scv[d * 2 + nt]);
    }

  int hi = lane >> 4, nl = lane & 15;
#pragma unroll
  for (int r = 0; r < 8; ++r) {
    int n = n0 + r + 8 * hi;
    size_t base = (size_t)n * kDIM;
#pragma unroll
    for (int nt = 0; nt < 4; ++nt) {
      int o = nt * 16 + nl;
      x16[base + o] = (_Float16)xs[nt][r];
      sc[base + o] = scs[nt][r];
    }
#pragma unroll
    for (int d = 0; d < 3; ++d)
#pragma unroll
      for (int t = 0; t < 2; ++t) {
        int o = t * 16 + nl;
        x16[base + 64 + o * 3 + d] = (_Float16)xv[d * 2 + t][r];
        sc[base + 64 + o * 3 + d] = scv[d * 2 + t][r];
      }
  }
}

// ---------------------------------------------------------------------------
// Kernel 2: edge tensor-product + gated MLP + scatter.  1 wave per 16 edges.
// ---------------------------------------------------------------------------
__global__ void __launch_bounds__(32)
edge_kernel(const float* __restrict__ edge_sh, const float* __restrict__ edge_fea,
            const float* __restrict__ elen, const int* __restrict__ edge_index,
            const _Float16* __restrict__ x16, const unsigned* __restrict__ wts,
            const float* __restrict__ fc_b0, const float* __restrict__ fc_b1,
            const float* __restrict__ fc_b2, float* __restrict__ node_acc,
            float* __restrict__ count) {
  __shared__ _Float16 s1_lds[16 * 192];    // [sa|sb|se]
  __shared__ _Float16 s1s_lds[16 * 192];   // s1 * s2[m]
  __shared__ _Float16 v1s_lds[3][16 * 96]; // v1_d * s2[m]
  __shared__ _Float16 vdot_lds[16 * 96];   // sum_d v1*v2
  __shared__ _Float16 el_lds[16 * 64];
  __shared__ _Float16 h_lds[16 * 64];
  __shared__ float sh_lds[16 * 4];
  __shared__ int ii_lds[16];
  __shared__ int jj_lds[16];

  int lane = threadIdx.x;
  int e0 = blockIdx.x * 16;

  if (lane < 16) {
    int e = e0 + lane;
    int ii = edge_index[e];
    int jj = edge_index[kE + e];
    ii_lds[lane] = ii; jj_lds[lane] = jj;
#pragma unroll
    for (int c = 0; c < 4; ++c) sh_lds[lane * 4 + c] = edge_sh[(size_t)e * 4 + c];
    atomicAddF(&count[ii], 1.0f);
  }
  if (lane == 0 && e0 + 16 < kE)
    __builtin_prefetch(edge_fea + (size_t)(e0 + 16) * kDIM, 0, 1);
  __syncthreads();

  int m = lane & 15, half = lane >> 4;
  {
    const _Float16* xi = x16 + (size_t)ii_lds[m] * kDIM;
    const _Float16* xj = x16 + (size_t)jj_lds[m] * kDIM;
    const float* ef = edge_fea + (size_t)(e0 + m) * kDIM;
    float s2f = sh_lds[m * 4 + 0];
    float sh1 = sh_lds[m * 4 + 1], sh2 = sh_lds[m * 4 + 2], sh3 = sh_lds[m * 4 + 3];
    _Float16 s2h = (_Float16)s2f;
    v8h s2v;
#pragma unroll
    for (int j = 0; j < 8; ++j) s2v[j] = s2h;

    // ---- s1 staging: 16B vector chunks, lane covers half a row ----
    v8h* d1 = (v8h*)(s1_lds + m * 192);
    v8h* d1s = (v8h*)(s1s_lds + m * 192);
    if (half == 0) {
      const v8h* si = (const v8h*)xi;
#pragma unroll
      for (int q = 0; q < 8; ++q) { v8h v = si[q]; d1[q] = v; d1s[q] = v * s2v; }
      const v8h* sj = (const v8h*)xj;
#pragma unroll
      for (int q = 0; q < 4; ++q) { v8h v = sj[q]; d1[8 + q] = v; d1s[8 + q] = v * s2v; }
    } else {
      const v8h* sj = (const v8h*)xj;
#pragma unroll
      for (int q = 0; q < 4; ++q) { v8h v = sj[4 + q]; d1[12 + q] = v; d1s[12 + q] = v * s2v; }
      const float4* ef4 = (const float4*)ef;
#pragma unroll
      for (int q = 0; q < 8; ++q) {
        v8h v = cvt8(ef4[2 * q], ef4[2 * q + 1]);
        d1[16 + q] = v; d1s[16 + q] = v * s2v;
      }
    }

    // ---- v1*s2 and vdot staging ----
    auto stageV = [&](int i, float a0, float a1, float a2) {
      v1s_lds[0][m * 96 + i] = (_Float16)(a0 * s2f);
      v1s_lds[1][m * 96 + i] = (_Float16)(a1 * s2f);
      v1s_lds[2][m * 96 + i] = (_Float16)(a2 * s2f);
      vdot_lds[m * 96 + i] = (_Float16)(a0 * sh1 + a1 * sh2 + a2 * sh3);
    };
    if (half == 0) {
#pragma unroll 4
      for (int i = 0; i < 32; ++i) {
        const _Float16* p = xi + 64 + i * 3;
        stageV(i, (float)p[0], (float)p[1], (float)p[2]);
      }
#pragma unroll 4
      for (int i = 32; i < 48; ++i) {
        const _Float16* p = xj + 64 + (i - 32) * 3;
        stageV(i, (float)p[0], (float)p[1], (float)p[2]);
      }
    } else {
#pragma unroll 4
      for (int i = 48; i < 64; ++i) {
        const _Float16* p = xj + 64 + (i - 32) * 3;
        stageV(i, (float)p[0], (float)p[1], (float)p[2]);
      }
#pragma unroll 4
      for (int i = 64; i < 96; ++i) {
        const float* p = ef + 64 + (i - 64) * 3;
        stageV(i, p[0], p[1], p[2]);
      }
    }

    // ---- elen staging ----
    const float4* el4 = (const float4*)(elen + (size_t)(e0 + m) * 64 + half * 32);
    v8h* dle = (v8h*)(el_lds + m * 64 + half * 32);
#pragma unroll
    for (int q = 0; q < 4; ++q) dle[q] = cvt8(el4[2 * q], el4[2 * q + 1]);
  }
  __syncthreads();

  int hi = half, nl = m;

  // ---- radial MLP: 64 -> 64 -> 64 -> 96 (K-outer) ----
  v8f h0[4] = {};
#pragma unroll
  for (int kt = 0; kt < 2; ++kt) {
    v16h a = load_a(el_lds, 64, kt * 32);
#pragma unroll
    for (int nt = 0; nt < 4; ++nt)
      h0[nt] = wmma16(a, load_b(wts + OFF_FC0, nt * 2 + kt), h0[nt]);
  }
#pragma unroll
  for (int nt = 0; nt < 4; ++nt) {
    float bb = fc_b0[nt * 16 + nl];
#pragma unroll
    for (int r = 0; r < 8; ++r)
      h_lds[(r + 8 * hi) * 64 + nt * 16 + nl] = (_Float16)siluf(h0[nt][r] + bb);
  }
  __syncthreads();
  v8f h1[4] = {};
#pragma unroll
  for (int kt = 0; kt < 2; ++kt) {
    v16h a = load_a(h_lds, 64, kt * 32);
#pragma unroll
    for (int nt = 0; nt < 4; ++nt)
      h1[nt] = wmma16(a, load_b(wts + OFF_FC1, nt * 2 + kt), h1[nt]);
  }
  __syncthreads();
#pragma unroll
  for (int nt = 0; nt < 4; ++nt) {
    float bb = fc_b1[nt * 16 + nl];
#pragma unroll
    for (int r = 0; r < 8; ++r)
      el_lds[(r + 8 * hi) * 64 + nt * 16 + nl] = (_Float16)siluf(h1[nt][r] + bb);
  }
  __syncthreads();
  v8f wacc[6] = {};
#pragma unroll
  for (int kt = 0; kt < 2; ++kt) {
    v16h a = load_a(el_lds, 64, kt * 32);
#pragma unroll
    for (int nt = 0; nt < 6; ++nt)
      wacc[nt] = wmma16(a, load_b(wts + OFF_FC2, nt * 2 + kt), wacc[nt]);
  }

  // ---- zs = s1*s2 @ Wss' + vdot @ Wvv' (K-outer, scales folded) ----
  v8f zs[6] = {};
#pragma unroll
  for (int kt = 0; kt < 6; ++kt) {
    v16h a = load_a(s1s_lds, 192, kt * 32);
#pragma unroll
    for (int nt = 0; nt < 6; ++nt)
      zs[nt] = wmma16(a, load_b(wts + OFF_WSS, nt * 6 + kt), zs[nt]);
  }
#pragma unroll
  for (int kt = 0; kt < 3; ++kt) {
    v16h a = load_a(vdot_lds, 96, kt * 32);
#pragma unroll
    for (int nt = 0; nt < 6; ++nt)
      zs[nt] = wmma16(a, load_b(wts + OFF_WVV, nt * 3 + kt), zs[nt]);
  }
  // ---- zsv = s1 @ Wsv' ----
  v8f zsv[2] = {};
#pragma unroll
  for (int kt = 0; kt < 6; ++kt) {
    v16h a = load_a(s1_lds, 192, kt * 32);
#pragma unroll
    for (int nt = 0; nt < 2; ++nt)
      zsv[nt] = wmma16(a, load_b(wts + OFF_WSV, nt * 6 + kt), zsv[nt]);
  }
  // ---- vs_d = (v1_d*s2) @ Wvs' ----
  v8f vs[6] = {};
#pragma unroll
  for (int d = 0; d < 3; ++d)
#pragma unroll
    for (int kt = 0; kt < 3; ++kt) {
      v16h a = load_a(v1s_lds[d], 96, kt * 32);
#pragma unroll
      for (int nt = 0; nt < 2; ++nt)
        vs[d * 2 + nt] = wmma16(a, load_b(wts + OFF_WVS, nt * 3 + kt), vs[d * 2 + nt]);
    }

  // ---- epilogue: gate, radial weight, scatter-add ----
  float b2s[4], b2v[2];
#pragma unroll
  for (int nt = 0; nt < 4; ++nt) b2s[nt] = fc_b2[nt * 16 + nl];
#pragma unroll
  for (int t = 0; t < 2; ++t) b2v[t] = fc_b2[64 + t * 16 + nl];

#pragma unroll
  for (int r = 0; r < 8; ++r) {
    int M = r + 8 * hi;
    float* nb = node_acc + (size_t)ii_lds[M] * kDIM;
#pragma unroll
    for (int nt = 0; nt < 4; ++nt) {
      int o = nt * 16 + nl;
      float os = siluf(zs[nt][r]) * (wacc[nt][r] + b2s[nt]);
      atomicAddF(&nb[o], os);
    }
    float v2d0 = sh_lds[M * 4 + 1];
    float v2d1 = sh_lds[M * 4 + 2];
    float v2d2 = sh_lds[M * 4 + 3];
#pragma unroll
    for (int t = 0; t < 2; ++t) {
      int o = t * 16 + nl;
      float g = 1.0f / (1.0f + __expf(-zs[4 + t][r]));
      float gw = g * (wacc[4 + t][r] + b2v[t]);
      float zb = zsv[t][r];
      atomicAddF(&nb[64 + o * 3 + 0], (zb * v2d0 + vs[0 + t][r]) * gw);
      atomicAddF(&nb[64 + o * 3 + 1], (zb * v2d1 + vs[2 + t][r]) * gw);
      atomicAddF(&nb[64 + o * 3 + 2], (zb * v2d2 + vs[4 + t][r]) * gw);
    }
  }
}

// ---------------------------------------------------------------------------
// Kernel 3a: node_out = e3linear(acc/cnt, Wpost) + sc ; accumulate batch sums
// ---------------------------------------------------------------------------
__global__ void __launch_bounds__(32)
node_post_kernel(const float* __restrict__ node_acc, const float* __restrict__ count,
                 const float* __restrict__ sc, const int* __restrict__ batch,
                 const unsigned* __restrict__ wts, float* __restrict__ node_pre,
                 float* __restrict__ stats) {
  __shared__ _Float16 sfeat[16 * 64];
  __shared__ _Float16 vfeat[3][16 * 32];
  __shared__ float inv_lds[16];
  __shared__ int b_lds[16];

  int lane = threadIdx.x;
  int n0 = blockIdx.x * 16;

  if (lane < 16) {
    int n = n0 + lane;
    inv_lds[lane] = 1.0f / fmaxf(count[n], 1.0f);
    int b = batch[n];
    b_lds[lane] = b;
    atomicAddF(&stats[b], 1.0f);  // cntb
  }
  __syncthreads();

  for (int t = lane; t < 16 * 160; t += 32) {
    int m = t / 160, c = t - m * 160;
    float val = node_acc[(size_t)(n0 + m) * kDIM + c] * inv_lds[m];
    if (c < 64) sfeat[m * 64 + c] = (_Float16)val;
    else {
      int i = (c - 64) / 3, d = (c - 64) - 3 * i;
      vfeat[d][m * 32 + i] = (_Float16)val;
    }
  }
  __syncthreads();

  v8f ps[4] = {}, pv[6] = {};
#pragma unroll
  for (int kt = 0; kt < 2; ++kt) {
    v16h a = load_a(sfeat, 64, kt * 32);
#pragma unroll
    for (int nt = 0; nt < 4; ++nt)
      ps[nt] = wmma16(a, load_b(wts + OFF_POSTS, nt * 2 + kt), ps[nt]);
  }
#pragma unroll
  for (int d = 0; d < 3; ++d) {
    v16h a = load_a(vfeat[d], 32, 0);
#pragma unroll
    for (int nt = 0; nt < 2; ++nt)
      pv[d * 2 + nt] = wmma16(a, load_b(wts + OFF_POSTV, nt), pv[d * 2 + nt]);
  }

  int hi = lane >> 4, nl = lane & 15;
  float ls[4] = {0.f, 0.f, 0.f, 0.f};
  float lv[12] = {0.f};
#pragma unroll
  for (int r = 0; r < 8; ++r) {
    int M = r + 8 * hi;
    int n = n0 + M;
    int b = b_lds[M];
    size_t base = (size_t)n * kDIM;
#pragma unroll
    for (int nt = 0; nt < 4; ++nt) {
      int o = nt * 16 + nl;
      float val = ps[nt][r] + sc[base + o];
      node_pre[base + o] = val;
      ls[b] += val;
    }
#pragma unroll
    for (int d = 0; d < 3; ++d)
#pragma unroll
      for (int t = 0; t < 2; ++t) {
        int o = t * 16 + nl;
        float val = pv[d * 2 + t][r] + sc[base + 64 + o * 3 + d];
        node_pre[base + 64 + o * 3 + d] = val;
        lv[b * 3 + d] += val;
      }
  }
#pragma unroll
  for (int b = 0; b < 4; ++b) atomicAddF(&stats[4 + b], ls[b]);
#pragma unroll
  for (int q = 0; q < 12; ++q) atomicAddF(&stats[8 + q], lv[q]);
}

// ---------------------------------------------------------------------------
// Kernel 3b: batch variance accumulation (one thread per node)
// ---------------------------------------------------------------------------
__global__ void var_kernel(const float* __restrict__ node_pre, const int* __restrict__ batch,
                           float* __restrict__ stats) {
  int n = blockIdx.x * blockDim.x + threadIdx.x;
  if (n >= kN) return;
  int b = batch[n];
  float cnt = fmaxf(stats[b], 1.0f);
  float ms = stats[4 + b] / (cnt * 64.0f);
  float mv0 = stats[8 + b * 3 + 0] / (cnt * 32.0f);
  float mv1 = stats[8 + b * 3 + 1] / (cnt * 32.0f);
  float mv2 = stats[8 + b * 3 + 2] / (cnt * 32.0f);
  const float* row = node_pre + (size_t)n * kDIM;
  float ss = 0.f, sv = 0.f;
#pragma unroll 8
  for (int i = 0; i < 64; ++i) { float c = row[i] - ms; ss += c * c; }
  for (int i = 0; i < 32; ++i) {
    float c0 = row[64 + i * 3 + 0] - mv0;
    float c1 = row[64 + i * 3 + 1] - mv1;
    float c2 = row[64 + i * 3 + 2] - mv2;
    sv += c0 * c0 + c1 * c1 + c2 * c2;
  }
  atomicAddF(&stats[20 + b], ss);
  atomicAddF(&stats[24 + b], sv);
}

// ---------------------------------------------------------------------------
// Kernel 3c: normalize + affine -> d_out
// ---------------------------------------------------------------------------
__global__ void norm_kernel(const float* __restrict__ node_pre, const int* __restrict__ batch,
                            const float* __restrict__ stats, const float* __restrict__ ln_w,
                            const float* __restrict__ ln_b, float* __restrict__ out) {
  int id = blockIdx.x * blockDim.x + threadIdx.x;
  if (id >= kN * kDIM) return;
  int n = id / kDIM, c = id - n * kDIM;
  int b = batch[n];
  float cnt = fmaxf(stats[b], 1.0f);
  float val = node_pre[id];
  if (c < 64) {
    float ms = stats[4 + b] / (cnt * 64.0f);
    float var = stats[20 + b] / (cnt * 64.0f);
    out[id] = (val - ms) / (sqrtf(var) + 1e-5f) * ln_w[c] + ln_b[c];
  } else {
    int i = (c - 64) / 3, d = (c - 64) - 3 * ((c - 64) / 3);
    float mv = stats[8 + b * 3 + d] / (cnt * 32.0f);
    float var = stats[24 + b] / (cnt * 96.0f);
    out[id] = (val - mv) / (sqrtf(var) + 1e-5f) * ln_w[64 + i];
  }
}

// ---------------------------------------------------------------------------
// Host launcher
// ---------------------------------------------------------------------------
extern "C" void kernel_launch(void* const* d_in, const int* in_sizes, int n_in,
                              void* d_out, int out_size, void* d_ws, size_t ws_size,
                              hipStream_t stream) {
  const float* node_fea = (const float*)d_in[0];
  const float* one_hot  = (const float*)d_in[1];
  const float* edge_sh  = (const float*)d_in[2];
  const float* edge_fea = (const float*)d_in[3];
  const float* elen     = (const float*)d_in[4];
  const int*   edge_idx = (const int*)d_in[5];
  const int*   batch    = (const int*)d_in[6];
  const float* Wpre_s   = (const float*)d_in[8];
  const float* Wpre_v   = (const float*)d_in[9];
  const float* Wss      = (const float*)d_in[10];
  const float* Wvv      = (const float*)d_in[11];
  const float* Wsv      = (const float*)d_in[12];
  const float* Wvs      = (const float*)d_in[13];
  const float* fc_w0    = (const float*)d_in[14];
  const float* fc_b0    = (const float*)d_in[15];
  const float* fc_w1    = (const float*)d_in[16];
  const float* fc_b1    = (const float*)d_in[17];
  const float* fc_w2    = (const float*)d_in[18];
  const float* fc_b2    = (const float*)d_in[19];
  const float* Wpost_s  = (const float*)d_in[20];
  const float* Wpost_v  = (const float*)d_in[21];
  const float* Wsc_s    = (const float*)d_in[22];
  const float* Wsc_v    = (const float*)d_in[23];
  const float* ln_w     = (const float*)d_in[24];
  const float* ln_b     = (const float*)d_in[25];

  char* ws = (char*)d_ws;
  unsigned* wts   = (unsigned*)ws;
  _Float16* x16   = (_Float16*)(ws + WOFF_X16);
  float* sc       = (float*)(ws + WOFF_SC);
  float* acc      = (float*)(ws + WOFF_ACC);
  float* cnt      = (float*)(ws + WOFF_CNT);
  float* pre      = (float*)(ws + WOFF_PRE);
  float* stats    = (float*)(ws + WOFF_STATS);

  hipMemsetAsync(acc, 0, (size_t)kN * kDIM * 4 + (size_t)kN * 4, stream);
  hipMemsetAsync(stats, 0, 64 * 4, stream);

  const float INV_SQRT3 = 0.57735026918962576f;   // 1/sqrt(3)
  const float INV_FAN   = 0.05892556509887896f;   // 1/sqrt(288)

  auto prep = [&](const float* W, int off, int K, int Nn, int ld, int tr, float scale) {
    int words = (K >> 5) * (Nn >> 4) * 256;
    prep_weights<<<(words + 255) / 256, 256, 0, stream>>>(W, wts + off, K, Nn, ld, tr, scale);
  };
  prep(Wss,    OFF_WSS,   192, 96, 192, 1, INV_FAN);
  prep(Wvv,    OFF_WVV,    96, 96,  96, 1, INV_FAN * INV_SQRT3);
  prep(Wsv,    OFF_WSV,   192, 32, 192, 1, INV_FAN);
  prep(Wvs,    OFF_WVS,    96, 32,  96, 1, INV_FAN);
  prep(fc_w0,  OFF_FC0,    64, 64,  64, 0, 1.0f);
  prep(fc_w1,  OFF_FC1,    64, 64,  64, 0, 1.0f);
  prep(fc_w2,  OFF_FC2,    64, 96,  96, 0, 1.0f);
  prep(Wpre_s, OFF_PRES,   64, 64,  64, 0, 0.125f);               // 1/sqrt(64)
  prep(Wpre_v, OFF_PREV,   32, 32,  32, 0, 0.17677669529663687f); // 1/sqrt(32)
  prep(Wpost_s,OFF_POSTS,  64, 64,  64, 0, 0.125f);
  prep(Wpost_v,OFF_POSTV,  32, 32,  32, 0, 0.17677669529663687f);
  prep(Wsc_s,  OFF_SCS,   256, 64, 256, 1, 0.0625f);              // 1/sqrt(256)
  prep(Wsc_v,  OFF_SCV,   128, 32, 128, 1, 0.08838834764831845f); // 1/sqrt(128)

  node_pre_kernel<<<kN / 16, 32, 0, stream>>>(node_fea, one_hot, wts, x16, sc);

  edge_kernel<<<kE / 16, 32, 0, stream>>>(edge_sh, edge_fea, elen, edge_idx, x16, wts,
                                          fc_b0, fc_b1, fc_b2, acc, cnt);

  node_post_kernel<<<kN / 16, 32, 0, stream>>>(acc, cnt, sc, batch, wts, pre, stats);

  var_kernel<<<(kN + 255) / 256, 256, 0, stream>>>(pre, batch, stats);

  norm_kernel<<<(kN * kDIM + 255) / 256, 256, 0, stream>>>(pre, batch, stats, ln_w, ln_b,
                                                           (float*)d_out);
}
